// Model_44633300140133
// MI455X (gfx1250) — matrix operand order repaired and checked by reading, not compiled, not run on hardware
//
#include <hip/hip_runtime.h>
#include <hip/hip_bf16.h>
#include <math.h>

typedef __bf16 bf16;
typedef __attribute__((ext_vector_type(16))) __bf16 v16bf;
typedef __attribute__((ext_vector_type(8)))  __bf16 v8bf;
typedef __attribute__((ext_vector_type(8)))  float  v8f;

#define DIMC 256
#define HEADS 8
#define HDIM 32
#define NEDGE 65536
#define NEL 131072
#define CLS_CHUNK 32768
#define TM 128
#define TN 64

// ---------------- elementwise helpers ----------------

__global__ void fill_f32_kernel(float* p, float v, int n) {
  int i = blockIdx.x * blockDim.x + threadIdx.x;
  if (i < n) p[i] = v;
}

// transpose f32 W[K,N] -> bf16 Wt[N,Kpad] (zero pad K..Kpad). batched via grid.z
__global__ void pack_wt_kernel(const float* W, bf16* Wt, int K, int N, int Kpad,
                               long sW, long sWt) {
  W  += (long)blockIdx.z * sW;
  Wt += (long)blockIdx.z * sWt;
  int i = blockIdx.x * blockDim.x + threadIdx.x;
  if (i >= N * Kpad) return;
  int n = i / Kpad, k = i % Kpad;
  float v = (k < K) ? W[(size_t)k * N + n] : 0.f;
  Wt[(size_t)n * Kpad + k] = (bf16)v;
}

// Conv2d(1,1,(4,4)) over [8192,235,4] -> bf16 [8192,256] (cols 232..255 pre-zeroed)
__global__ void conv_kernel(const float* __restrict__ x, const float* __restrict__ w,
                            const float* __restrict__ b, bf16* out) {
  int i = blockIdx.x * blockDim.x + threadIdx.x;
  if (i >= 8192 * 232) return;
  int n = i / 232, t = i % 232;
  const float* xp = x + (size_t)n * 235 * 4 + t * 4;
  float acc = b[0];
#pragma unroll
  for (int kh = 0; kh < 4; kh++)
#pragma unroll
    for (int kw = 0; kw < 4; kw++)
      acc += xp[kh * 4 + kw] * w[kh * 4 + kw];
  out[(size_t)n * 256 + t] = (bf16)acc;
}

__global__ void gelu_bf16_kernel(const float* in, bf16* out, int n) {
  int i = blockIdx.x * blockDim.x + threadIdx.x;
  if (i >= n) return;
  float x = in[i];
  out[i] = (bf16)(0.5f * x * (1.f + erff(x * 0.70710678118654752f)));
}

__global__ void skip_blend_kernel(const float* o, const float* skip, float* xf,
                                  bf16* xb, int n) {
  int i = blockIdx.x * blockDim.x + threadIdx.x;
  if (i >= n) return;
  float s = 1.f / (1.f + expf(-skip[0]));
  float v = s * o[i] + (1.f - s) * xf[i];
  xf[i] = v;
  xb[i] = (bf16)v;
}

__global__ void gather_concat_kernel(const int* eli, const bf16* xm, const bf16* xd,
                                     bf16* hcat, int base, int rows) {
  int i = blockIdx.x * blockDim.x + threadIdx.x;
  if (i >= rows * 512) return;
  int row = i >> 9, col = i & 511;
  int ge = base + row;
  bf16 v = (col < 256) ? xm[(size_t)eli[ge] * 256 + col]
                       : xd[(size_t)eli[NEL + ge] * 256 + (col - 256)];
  hcat[(size_t)row * 512 + col] = v;
}

// ------------- simple WMMA GEMM (one wave = one 16x16 tile, batched) -------------
// flags: bit0 write f32 C, bit1 write bf16 C, bit2 A is f32 (convert on the fly)
__global__ void gemm_wmma_kernel(const void* Av, int lda, long sA,
                                 const bf16* __restrict__ B, int ldb, long sB,
                                 const float* __restrict__ bias,
                                 float* Cf, bf16* Cb, int ldc, long sC,
                                 int M, int N, int K, int flags) {
  long z = blockIdx.z;
  const bf16*  Ab = (const bf16*)Av + z * sA;
  const float* Af = (const float*)Av + z * sA;
  B += z * sB;
  int wave = threadIdx.x >> 5;
  int lane = threadIdx.x & 31;
  int tn = N >> 4;
  int tile = blockIdx.x * (blockDim.x >> 5) + wave;
  int tiles = (M >> 4) * tn;
  if (tile >= tiles) return;
  int m0 = (tile / tn) << 4;
  int n0 = (tile % tn) << 4;
  int half = lane >> 4, r = lane & 15;

  v8f c = {0.f, 0.f, 0.f, 0.f, 0.f, 0.f, 0.f, 0.f};
  for (int k0 = 0; k0 < K; k0 += 32) {
    v16bf av, bv;
    if (flags & 4) {
      const float* pa = Af + (size_t)(m0 + r) * lda + k0 + half * 8;
#pragma unroll
      for (int j = 0; j < 8; j++) av[j] = (bf16)pa[j];
#pragma unroll
      for (int j = 0; j < 8; j++) av[j + 8] = (bf16)pa[16 + j];
    } else {
      const bf16* pa = Ab + (size_t)(m0 + r) * lda + k0 + half * 8;
      *(v8bf*)&av       = *(const v8bf*)pa;
      *((v8bf*)&av + 1) = *(const v8bf*)(pa + 16);
    }
    const bf16* pb = B + (size_t)(n0 + r) * ldb + k0 + half * 8;
    *(v8bf*)&bv       = *(const v8bf*)pb;
    *((v8bf*)&bv + 1) = *(const v8bf*)(pb + 16);
    c = __builtin_amdgcn_wmma_f32_16x16x32_bf16(false, av, false, bv,
                                                (short)0, c, false, false);
  }
  float bv0 = bias ? bias[n0 + r] : 0.f;
  int row0 = m0 + half * 8;
#pragma unroll
  for (int i = 0; i < 8; i++) {
    float val = c[i] + bv0;
    size_t idx = (size_t)(row0 + i) * ldc + (n0 + r);
    if (flags & 1) (Cf + z * sC)[idx] = val;
    if (flags & 2) (Cb + z * sC)[idx] = (bf16)val;
  }
}

// ------------- tiled WMMA GEMM: block = 8 waves, tile 128x64 -------------
// Each wave: rows [m0b + wave*16, +16), all 64 cols -> 4 accumulators; the A
// fragment is loaded once per K-step and reused across 4 WMMAs. B tile (64x32
// bf16 = 4KB) is staged in LDS once per K-step and shared by all 8 waves.
// Requires M%128==0, N%64==0, K%32==0.
__global__ void gemm_wmma_tiled_kernel(const void* Av, int lda,
                                       const bf16* __restrict__ B, int ldb,
                                       const float* __restrict__ bias,
                                       float* Cf, bf16* Cb, int ldc,
                                       int M, int N, int K, int flags) {
  __shared__ __align__(16) bf16 lb[TN * 32];
  int nb = N / TN;
  int m0b = (blockIdx.x / nb) * TM;
  int n0b = (blockIdx.x % nb) * TN;
  int wave = threadIdx.x >> 5, lane = threadIdx.x & 31;
  int half = lane >> 4, r = lane & 15;
  int mrow = m0b + wave * 16 + r;
  const float* Af = (const float*)Av;
  const bf16*  Ab = (const bf16*)Av;

  v8f acc[4];
#pragma unroll
  for (int g = 0; g < 4; g++)
    acc[g] = (v8f){0.f, 0.f, 0.f, 0.f, 0.f, 0.f, 0.f, 0.f};

  int srow = threadIdx.x >> 1;          // staging: 128 threads, 16B each
  int sseg = threadIdx.x & 1;

  for (int k0 = 0; k0 < K; k0 += 32) {
    __syncthreads();
    if (threadIdx.x < 128) {
      const uint4* src = (const uint4*)(B + (size_t)(n0b + srow) * ldb + k0 + sseg * 16);
      *(uint4*)&lb[srow * 32 + sseg * 16] = *src;
      if (k0 + 32 < K)                   // read-ahead next B K-step
        __builtin_prefetch(B + (size_t)(n0b + srow) * ldb + k0 + 32 + sseg * 16, 0, 3);
    }
    v16bf av;
    if (flags & 4) {
      const float* pa = Af + (size_t)mrow * lda + k0 + half * 8;
#pragma unroll
      for (int j = 0; j < 8; j++) av[j] = (bf16)pa[j];
#pragma unroll
      for (int j = 0; j < 8; j++) av[j + 8] = (bf16)pa[16 + j];
    } else {
      const bf16* pa = Ab + (size_t)mrow * lda + k0 + half * 8;
      *(v8bf*)&av       = *(const v8bf*)pa;
      *((v8bf*)&av + 1) = *(const v8bf*)(pa + 16);
      if (k0 + 32 < K)
        __builtin_prefetch(pa + 32, 0, 3);
    }
    __syncthreads();
#pragma unroll
    for (int g = 0; g < 4; g++) {
      const bf16* pb = &lb[(g * 16 + r) * 32 + half * 8];
      v16bf bv;
      *(v8bf*)&bv       = *(const v8bf*)pb;
      *((v8bf*)&bv + 1) = *(const v8bf*)(pb + 16);
      acc[g] = __builtin_amdgcn_wmma_f32_16x16x32_bf16(false, av, false, bv,
                                                       (short)0, acc[g], false, false);
    }
  }
  int row0 = m0b + wave * 16 + half * 8;
#pragma unroll
  for (int g = 0; g < 4; g++) {
    int col = n0b + g * 16 + r;
    float bv0 = bias ? bias[col] : 0.f;
#pragma unroll
    for (int i = 0; i < 8; i++) {
      float val = acc[g][i] + bv0;
      size_t idx = (size_t)(row0 + i) * ldc + col;
      if (flags & 1) Cf[idx] = val;
      if (flags & 2) Cb[idx] = (bf16)val;
    }
  }
}

// ---------------- edge attention ----------------

__device__ __forceinline__ void atomicMaxF32(float* addr, float val) {
  if (val >= 0.f) atomicMax((int*)addr, __float_as_int(val));
  else            atomicMin((unsigned int*)addr, __float_as_uint(val));
}
__device__ __forceinline__ void atomAddF32(float* p, float v) {
  __hip_atomic_fetch_add(p, v, __ATOMIC_RELAXED, __HIP_MEMORY_SCOPE_AGENT);
}

// one wave per edge: alpha[e,h] = dot(q[dst,h,:], kk[src,h,:]) * p_rel[h] / sqrt(32)
__global__ void edge_alpha_kernel(const int* __restrict__ ei, const float* __restrict__ q,
                                  const float* __restrict__ kk, const float* __restrict__ prel,
                                  float* alpha, float* amax, int E) {
  int gw = (blockIdx.x * blockDim.x + threadIdx.x) >> 5;
  if (gw >= E) return;
  int lane = threadIdx.x & 31;
  int head = lane >> 2, sub = lane & 3;
  int src = ei[gw], dst = ei[E + gw];
  const float* qp = q  + (size_t)dst * DIMC + head * HDIM + sub * 8;
  const float* kp = kk + (size_t)src * DIMC + head * HDIM + sub * 8;
  float p = 0.f;
#pragma unroll
  for (int j = 0; j < 8; j++) p += qp[j] * kp[j];
  p += __shfl_xor(p, 1, 32);
  p += __shfl_xor(p, 2, 32);
  if (sub == 0) {
    float v = p * prel[head] * 0.17677669529663687f; // 1/sqrt(32)
    alpha[(size_t)gw * HEADS + head] = v;
    atomicMaxF32(&amax[(size_t)dst * HEADS + head], v);
  }
}

__global__ void edge_exp_kernel(const int* __restrict__ ei, float* alpha,
                                const float* __restrict__ amax, float* esum, int E) {
  int i = blockIdx.x * blockDim.x + threadIdx.x;
  if (i >= E * HEADS) return;
  int e = i >> 3, h = i & 7;
  int dst = ei[E + e];
  float ex = expf(alpha[i] - amax[(size_t)dst * HEADS + h]);
  alpha[i] = ex;
  atomAddF32(&esum[(size_t)dst * HEADS + h], ex);
}

__global__ void edge_scatter_kernel(const int* __restrict__ ei, const float* __restrict__ alpha,
                                    const float* __restrict__ esum, const float* __restrict__ vv,
                                    float* out, int E) {
  int gw = (blockIdx.x * blockDim.x + threadIdx.x) >> 5;
  if (gw >= E) return;
  int lane = threadIdx.x & 31;
  int head = lane >> 2, sub = lane & 3;
  int src = ei[gw], dst = ei[E + gw];
  float w = alpha[(size_t)gw * HEADS + head] /
            (esum[(size_t)dst * HEADS + head] + 1e-16f);
  const float* vp = vv + (size_t)src * DIMC + head * HDIM + sub * 8;
  float* op = out + (size_t)dst * DIMC + head * HDIM + sub * 8;
#pragma unroll
  for (int j = 0; j < 8; j++) atomAddF32(&op[j], vp[j] * w);
}

// one wave per row: out = sigmoid(dot(h1[row], w2) + b2)
__global__ void cls_final_kernel(const float* __restrict__ h1, const float* __restrict__ w2,
                                 const float* __restrict__ b2, float* out, int base, int rows) {
  int gw = (blockIdx.x * blockDim.x + threadIdx.x) >> 5;
  if (gw >= rows) return;
  int lane = threadIdx.x & 31;
  const float* hp = h1 + (size_t)gw * 256 + lane * 8;
  const float* wp = w2 + lane * 8;
  float p = 0.f;
#pragma unroll
  for (int j = 0; j < 8; j++) p += hp[j] * wp[j];
#pragma unroll
  for (int off = 16; off; off >>= 1) p += __shfl_xor(p, off, 32);
  if (lane == 0) out[base + gw] = 1.f / (1.f + expf(-(p + b2[0])));
}

// ---------------- host orchestration ----------------
// Input leaf order: top-level dict insertion order; `params` pytree flattened with
// JAX semantics (dict keys alphabetical). 133 leaves total.
extern "C" void kernel_launch(void* const* d_in, const int* in_sizes, int n_in,
                              void* d_out, int out_size, void* d_ws, size_t ws_size,
                              hipStream_t stream) {
  if (n_in < 133) return;
  const int Nn[3] = {8192, 8192, 16384};       // mirna, disease, mrna
  const int NTP[3] = {1, 0, 2};                // node type -> alphabetical param slot
  struct ETI { int ein, srct, dstt, es; };
  const ETI ET[9] = {                          // es = alphabetical edge index
    {3, 0, 0, 2},  // ei_family
    {4, 1, 1, 3},  // ei_fatherson
    {5, 0, 2, 6},  // ei_mirna_mrna
    {6, 2, 0, 8},  // ei_mrna_mirna
    {7, 2, 1, 7},  // ei_mrna_disease
    {8, 1, 2, 1},  // ei_disease_mrna
    {9, 2, 2, 4},  // ei_group
    {10, 0, 1, 5}, // ei_mirna_disease
    {11, 1, 0, 0}, // ei_disease_mirna
  };
  const int IDX_CLS_B1 = 13, IDX_CLS_B2 = 14, IDX_CLS_W1 = 15, IDX_CLS_W2 = 16,
            IDX_CONV_B = 17, IDX_CONV_W = 18, IDX_DEMB_B = 19, IDX_DEMB_W = 20,
            IDX_L0 = 21, LSTRIDE = 54,
            IDX_MEMB_B = 129, IDX_MEMB_W = 130, IDX_REMB_B = 131, IDX_REMB_W = 132;
  auto P = [&](int i) { return (const float*)d_in[i]; };

  char* ws = (char*)d_ws;
  size_t off = 0;
  auto alloc = [&](size_t bytes) -> char* {
    size_t o = (off + 255) & ~(size_t)255;
    off = o + bytes;
    return ws + o;
  };

  float *xf[3], *qf[3], *outacc[3];
  bf16 *xb[3], *kb[3], *vb[3];
  for (int t = 0; t < 3; t++) {
    xf[t]     = (float*)alloc((size_t)Nn[t] * DIMC * 4);
    qf[t]     = (float*)alloc((size_t)Nn[t] * DIMC * 4);
    outacc[t] = (float*)alloc((size_t)Nn[t] * DIMC * 4);
    xb[t]     = (bf16*)alloc((size_t)Nn[t] * DIMC * 2);
    kb[t]     = (bf16*)alloc((size_t)Nn[t] * DIMC * 2);
    vb[t]     = (bf16*)alloc((size_t)Nn[t] * DIMC * 2);
  }
  float* kkbuf = (float*)alloc((size_t)16384 * DIMC * 4);
  float* vvbuf = (float*)alloc((size_t)16384 * DIMC * 4); // reused as a-proj output
  float* alpha = (float*)alloc((size_t)NEDGE * HEADS * 4);
  float* amax  = (float*)alloc((size_t)16384 * HEADS * 4);
  float* esum  = (float*)alloc((size_t)16384 * HEADS * 4);
  float* h1    = (float*)alloc((size_t)CLS_CHUNK * DIMC * 4);
  bf16* geb    = (bf16*)alloc((size_t)16384 * DIMC * 2);
  bf16* conv_a = (bf16*)alloc((size_t)8192 * 256 * 2);
  bf16* hcat   = (bf16*)alloc((size_t)CLS_CHUNK * 512 * 2);
  bf16* wt_memb = (bf16*)alloc((size_t)256 * 256 * 2);
  bf16* wt_demb = (bf16*)alloc((size_t)256 * 1536 * 2);
  bf16* wt_remb = (bf16*)alloc((size_t)256 * 1536 * 2);
  bf16* wt_cls1 = (bf16*)alloc((size_t)256 * 512 * 2);
  bf16 *wt_k[2][3], *wt_q[2][3], *wt_v[2][3], *wt_a[2][3];
  bf16 *wt_ar[2][9], *wt_mr[2][9];
  for (int l = 0; l < 2; l++) {
    for (int t = 0; t < 3; t++) {
      wt_k[l][t] = (bf16*)alloc(256 * 256 * 2);
      wt_q[l][t] = (bf16*)alloc(256 * 256 * 2);
      wt_v[l][t] = (bf16*)alloc(256 * 256 * 2);
      wt_a[l][t] = (bf16*)alloc(256 * 256 * 2);
    }
    for (int e = 0; e < 9; e++) {
      wt_ar[l][e] = (bf16*)alloc(8 * 32 * 32 * 2);
      wt_mr[l][e] = (bf16*)alloc(8 * 32 * 32 * 2);
    }
  }

  auto pack = [&](const float* W, bf16* Wt, int K, int N, int Kpad,
                  int batch, long sW, long sWt) {
    dim3 grid((N * Kpad + 255) / 256, 1, batch);
    pack_wt_kernel<<<grid, 256, 0, stream>>>(W, Wt, K, N, Kpad, sW, sWt);
  };
  // dispatch: batched / irregular -> simple kernel; big regular -> tiled kernel
  auto gemm = [&](const void* A, int lda, const bf16* B, int ldb, const float* bias,
                  float* Cf, bf16* Cb, int ldc, int M, int N, int K, int flags,
                  int batch, long sA, long sB, long sC) {
    if (batch == 1 && (M % TM) == 0 && (N % TN) == 0) {
      dim3 grid((M / TM) * (N / TN));
      gemm_wmma_tiled_kernel<<<grid, 256, 0, stream>>>(A, lda, B, ldb, bias,
                                                       Cf, Cb, ldc, M, N, K, flags);
    } else {
      int tiles = (M / 16) * (N / 16);
      dim3 grid((tiles + 7) / 8, 1, batch);
      gemm_wmma_kernel<<<grid, 256, 0, stream>>>(A, lda, sA, B, ldb, sB, bias,
                                                 Cf, Cb, ldc, sC, M, N, K, flags);
    }
  };
  auto fillf = [&](float* p, float v, int n) {
    fill_f32_kernel<<<(n + 255) / 256, 256, 0, stream>>>(p, v, n);
  };

  // ---- pack all weights (bf16, transposed) ----
  pack(P(IDX_MEMB_W), wt_memb, 232, 256, 256, 1, 0, 0);
  pack(P(IDX_DEMB_W), wt_demb, 1536, 256, 1536, 1, 0, 0);
  pack(P(IDX_REMB_W), wt_remb, 1536, 256, 1536, 1, 0, 0);
  pack(P(IDX_CLS_W1), wt_cls1, 512, 256, 512, 1, 0, 0);
  for (int l = 0; l < 2; l++) {
    int L = IDX_L0 + l * LSTRIDE;
    for (int t = 0; t < 3; t++) {
      int p2 = NTP[t] * 2;
      pack(P(L + 0 + p2 + 1),  wt_a[l][t], 256, 256, 256, 1, 0, 0);
      pack(P(L + 15 + p2 + 1), wt_k[l][t], 256, 256, 256, 1, 0, 0);
      pack(P(L + 39 + p2 + 1), wt_q[l][t], 256, 256, 256, 1, 0, 0);
      pack(P(L + 48 + p2 + 1), wt_v[l][t], 256, 256, 256, 1, 0, 0);
    }
    for (int e = 0; e < 9; e++) {
      pack(P(L + 6 + e),  wt_ar[l][e], 32, 32, 32, 8, 1024, 1024);
      pack(P(L + 21 + e), wt_mr[l][e], 32, 32, 32, 8, 1024, 1024);
    }
  }

  // ---- embeddings ----
  hipMemsetAsync(conv_a, 0, (size_t)8192 * 256 * 2, stream);
  conv_kernel<<<(8192 * 232 + 255) / 256, 256, 0, stream>>>(
      (const float*)d_in[0], P(IDX_CONV_W), P(IDX_CONV_B), conv_a);
  gemm(conv_a, 256, wt_memb, 256, P(IDX_MEMB_B), xf[0], xb[0], 256,
       8192, 256, 256, 3, 1, 0, 0, 0);
  gemm(d_in[1], 1536, wt_demb, 1536, P(IDX_DEMB_B), xf[1], xb[1], 256,
       8192, 256, 1536, 7, 1, 0, 0, 0);
  gemm(d_in[2], 1536, wt_remb, 1536, P(IDX_REMB_B), xf[2], xb[2], 256,
       16384, 256, 1536, 7, 1, 0, 0, 0);

  // ---- HGT layers ----
  for (int l = 0; l < 2; l++) {
    int L = IDX_L0 + l * LSTRIDE;
    for (int t = 0; t < 3; t++) {
      int p2 = NTP[t] * 2;
      gemm(xb[t], 256, wt_k[l][t], 256, P(L + 15 + p2), nullptr, kb[t], 256,
           Nn[t], 256, 256, 2, 1, 0, 0, 0);
      gemm(xb[t], 256, wt_q[l][t], 256, P(L + 39 + p2), qf[t], nullptr, 256,
           Nn[t], 256, 256, 1, 1, 0, 0, 0);
      gemm(xb[t], 256, wt_v[l][t], 256, P(L + 48 + p2), nullptr, vb[t], 256,
           Nn[t], 256, 256, 2, 1, 0, 0, 0);
      hipMemsetAsync(outacc[t], 0, (size_t)Nn[t] * DIMC * 4, stream);
    }
    for (int e = 0; e < 9; e++) {
      const ETI& et = ET[e];
      int Nsrc = Nn[et.srct], Ndst = Nn[et.dstt];
      const int* ei = (const int*)d_in[et.ein];
      fillf(amax, -INFINITY, Ndst * HEADS);
      hipMemsetAsync(esum, 0, (size_t)Ndst * HEADS * 4, stream);
      // kk = einsum('nhd,hde->nhe', k, a_rel): batched per-head 32x32 GEMM
      gemm(kb[et.srct], 256, wt_ar[l][et.es], 32, nullptr, kkbuf, nullptr, 256,
           Nsrc, 32, 32, 1, 8, 32, 1024, 32);
      gemm(vb[et.srct], 256, wt_mr[l][et.es], 32, nullptr, vvbuf, nullptr, 256,
           Nsrc, 32, 32, 1, 8, 32, 1024, 32);
      edge_alpha_kernel<<<(NEDGE * 32) / 256, 256, 0, stream>>>(
          ei, qf[et.dstt], kkbuf, P(L + 30 + et.es), alpha, amax, NEDGE);
      edge_exp_kernel<<<(NEDGE * 8) / 256, 256, 0, stream>>>(ei, alpha, amax, esum, NEDGE);
      edge_scatter_kernel<<<(NEDGE * 32) / 256, 256, 0, stream>>>(
          ei, alpha, esum, vvbuf, outacc[et.dstt], NEDGE);
    }
    for (int t = 0; t < 3; t++) {
      int p2 = NTP[t] * 2;
      int n = Nn[t] * DIMC;
      gelu_bf16_kernel<<<(n + 255) / 256, 256, 0, stream>>>(outacc[t], geb, n);
      gemm(geb, 256, wt_a[l][t], 256, P(L + 0 + p2), vvbuf, nullptr, 256,
           Nn[t], 256, 256, 1, 1, 0, 0, 0);
      skip_blend_kernel<<<(n + 255) / 256, 256, 0, stream>>>(
          vvbuf, P(L + 45 + NTP[t]), xf[t], xb[t], n);
    }
  }

  // ---- classifier (chunked) ----
  const int* eli = (const int*)d_in[12];
  for (int c = 0; c < NEL / CLS_CHUNK; c++) {
    int base = c * CLS_CHUNK;
    gather_concat_kernel<<<(CLS_CHUNK * 512) / 256, 256, 0, stream>>>(
        eli, xb[0], xb[1], hcat, base, CLS_CHUNK);
    gemm(hcat, 512, wt_cls1, 512, P(IDX_CLS_B1), h1, nullptr, 256,
         CLS_CHUNK, 256, 512, 1, 1, 0, 0, 0);
    cls_final_kernel<<<(CLS_CHUNK * 32) / 256, 256, 0, stream>>>(
        h1, P(IDX_CLS_W2), P(IDX_CLS_B2), (float*)d_out, base, CLS_CHUNK);
  }
  (void)in_sizes; (void)out_size; (void)ws_size;
}
